// AOGStructure_74792560493275
// MI455X (gfx1250) — matrix operand order repaired and checked by reading, not compiled, not run on hardware
//
#include <hip/hip_runtime.h>
#include <hip/hip_bf16.h>

typedef __attribute__((ext_vector_type(16))) __bf16       v16bf;
typedef __attribute__((ext_vector_type(8)))  float        v8f;
typedef __attribute__((ext_vector_type(8)))  unsigned int v8u;
typedef __attribute__((ext_vector_type(4)))  unsigned int v4u;

// f32 -> bf16 (RNE). Let clang pick the native gfx1250 lowering
// (ideally v_cvt_pk_bf16_f32) instead of a manual 3-op sequence.
__device__ __forceinline__ unsigned short f2bu(float f) {
    return __builtin_bit_cast(unsigned short, static_cast<__bf16>(f));
}
__device__ __forceinline__ __bf16 f2b(float f) {
    return static_cast<__bf16>(f);
}

// ---------------------------------------------------------------------------
// Generic wave-tiled GEMM: C[M,N] = A[M,K] * B[K,N] (+bias), bf16 WMMA, f32 acc.
// B is supplied transposed (Bt: N x K, bf16) so each lane's B-fragment is a
// contiguous 32-byte load. Each wave owns a 16x64 tile: one A fragment feeds
// four independent v_wmma_f32_16x16x32_bf16 per K-step.
// ---------------------------------------------------------------------------
template <bool AF32>
__global__ void __launch_bounds__(256)
gemm_bf16_wmma(const void* __restrict__ Asrc, int lda,
               const unsigned short* __restrict__ Bt,   // N x K (bf16 bits)
               const float* __restrict__ bias,          // length N or nullptr
               float* __restrict__ Cf,                   // f32 out or nullptr
               unsigned short* __restrict__ Cb,          // bf16 out or nullptr
               int ldc, int M, int N, int K)
{
    const int gwave = (blockIdx.x * 256 + threadIdx.x) >> 5;
    const int lane  = threadIdx.x & 31;
    const int ntiles = N >> 6;                 // 64-wide wave tiles
    const int mt = gwave / ntiles;
    const int nt = gwave - mt * ntiles;
    if (mt * 16 >= M) return;

    const int half = lane >> 4;                // 0: lanes 0-15, 1: lanes 16-31
    const int l16  = lane & 15;
    const int row  = mt * 16 + l16;            // A-matrix row for this lane

    v8f acc[4] = {v8f{}, v8f{}, v8f{}, v8f{}};

    const unsigned short* bptr[4];
#pragma unroll
    for (int t = 0; t < 4; ++t)
        bptr[t] = Bt + (size_t)(nt * 64 + t * 16 + l16) * K + half * 16;

    for (int k = 0; k < K; k += 32) {
        v16bf afrag;
        if (AF32) {
            const float* Af = (const float*)Asrc + (size_t)row * lda + k + half * 16;
            v8f lo = *(const v8f*)(Af);        // K = k+half*16 .. +7
            v8f hi = *(const v8f*)(Af + 8);    // K = k+half*16+8 .. +15
#pragma unroll
            for (int i = 0; i < 8; ++i) {
                afrag[i]     = f2b(lo[i]);
                afrag[8 + i] = f2b(hi[i]);
            }
        } else {
            const unsigned short* Ab =
                (const unsigned short*)Asrc + (size_t)row * lda + k + half * 16;
            v4u lo = *(const v4u*)(Ab);
            v4u hi = *(const v4u*)(Ab + 8);
            v8u a;
#pragma unroll
            for (int i = 0; i < 4; ++i) { a[i] = lo[i]; a[4 + i] = hi[i]; }
            afrag = __builtin_bit_cast(v16bf, a);
        }
#pragma unroll
        for (int t = 0; t < 4; ++t) {
            v16bf bfrag = __builtin_bit_cast(v16bf, *(const v8u*)(bptr[t] + k));
            acc[t] = __builtin_amdgcn_wmma_f32_16x16x32_bf16(
                false, afrag, false, bfrag, (short)0, acc[t], false, false);
        }
    }

#pragma unroll
    for (int t = 0; t < 4; ++t) {
        const int c  = nt * 64 + t * 16 + l16;
        const float bv = bias ? bias[c] : 0.0f;
#pragma unroll
        for (int d = 0; d < 8; ++d) {
            const int r = mt * 16 + half * 8 + d;   // C/D VGPR layout
            float v = acc[t][d] + bv;
            if (Cf) Cf[(size_t)r * ldc + c] = v;
            if (Cb) Cb[(size_t)r * ldc + c] = f2bu(v);
        }
    }
}

// ---------------------------------------------------------------------------
// Streaming f32 -> bf16 conversion (8 elements per thread: 32 B in, b128 out).
// ---------------------------------------------------------------------------
__global__ void convert_bf16(const float* __restrict__ src,
                             unsigned short* __restrict__ dst, int n8)
{
    for (int i = blockIdx.x * blockDim.x + threadIdx.x; i < n8;
         i += gridDim.x * blockDim.x) {
        v8f v = *(const v8f*)(src + (size_t)i * 8);
        v4u o;
#pragma unroll
        for (int j = 0; j < 4; ++j)
            o[j] = (unsigned)f2bu(v[2 * j]) | ((unsigned)f2bu(v[2 * j + 1]) << 16);
        *(v4u*)(dst + (size_t)i * 8) = o;
    }
}

// ---------------------------------------------------------------------------
// One-time weight transpose + f32->bf16 conversion, bias concatenation.
// ---------------------------------------------------------------------------
__global__ void prep_kernel(const float* __restrict__ Wp,  const float* __restrict__ bp,
                            const float* __restrict__ Wpr, const float* __restrict__ bpr,
                            const float* __restrict__ Wo,
                            const float* __restrict__ Wm_obj, const float* __restrict__ bm_obj,
                            const float* __restrict__ Wm_per, const float* __restrict__ bm_per,
                            unsigned short* __restrict__ W1t,    // 512 x 2048
                            unsigned short* __restrict__ Wot,    // 256 x 2048
                            unsigned short* __restrict__ WmTopT, // 512 x 256
                            unsigned short* __restrict__ WmOBot, // 256 x 256
                            unsigned short* __restrict__ WmPBot, // 256 x 256
                            float* __restrict__ bias1,           // 512
                            float* __restrict__ biasTop)         // 512
{
    const int total = 512 * 2048;
    for (int i = blockIdx.x * blockDim.x + threadIdx.x; i < total;
         i += gridDim.x * blockDim.x) {
        {   // W1t = [Wp | Wpr]^T
            int n = i >> 11, k = i & 2047;
            float v = (n < 256) ? Wp[k * 256 + n] : Wpr[k * 256 + (n - 256)];
            W1t[(size_t)n * 2048 + k] = f2bu(v);
        }
        if (i < 256 * 2048) {
            int n = i >> 11, k = i & 2047;
            Wot[(size_t)n * 2048 + k] = f2bu(Wo[k * 256 + n]);
        }
        if (i < 512 * 256) {
            int n = i >> 8, k = i & 255;
            float v = (n < 256) ? Wm_obj[k * 256 + n] : Wm_per[k * 256 + (n - 256)];
            WmTopT[n * 256 + k] = f2bu(v);
        }
        if (i < 256 * 256) {
            int n = i >> 8, k = i & 255;
            WmOBot[n * 256 + k] = f2bu(Wm_obj[(256 + k) * 256 + n]);
            WmPBot[n * 256 + k] = f2bu(Wm_per[(256 + k) * 256 + n]);
        }
        if (i < 512) {
            bias1[i]   = (i < 256) ? bp[i]     : bpr[i - 256];
            biasTop[i] = (i < 256) ? bm_obj[i] : bm_per[i - 256];
        }
    }
}

// ---------------------------------------------------------------------------
// objs_interact: out[f,i,c] += max_{j<48} b_o[f,j,c]
// ---------------------------------------------------------------------------
__global__ void reduce_obj(const float* __restrict__ b_o, float* __restrict__ out)
{
    int t = blockIdx.x * blockDim.x + threadIdx.x;   // 128 frames * 256 channels
    if (t >= 128 * 256) return;
    int f = t >> 8, c = t & 255;
    const float* p = b_o + (size_t)f * 48 * 256 + c;
    float m = p[0];
#pragma unroll 4
    for (int j = 1; j < 48; ++j) m = fmaxf(m, p[(size_t)j * 256]);
    float* o = out + (size_t)f * 16 * 512 + c;
    for (int i = 0; i < 16; ++i) o[(size_t)i * 512] += m;
}

// ---------------------------------------------------------------------------
// persons_interact (diagonal masked): top-2 max over j, use max2 when i==argmax
// ---------------------------------------------------------------------------
__global__ void reduce_per(const float* __restrict__ b_p, float* __restrict__ out)
{
    int t = blockIdx.x * blockDim.x + threadIdx.x;
    if (t >= 128 * 256) return;
    int f = t >> 8, c = t & 255;
    const float* p = b_p + (size_t)f * 16 * 256 + c;
    float m1 = -__builtin_inff(), m2 = -__builtin_inff();
    int j1 = -1;
#pragma unroll
    for (int j = 0; j < 16; ++j) {
        float v = p[(size_t)j * 256];
        if (v > m1) { m2 = m1; m1 = v; j1 = j; }
        else if (v > m2) { m2 = v; }
    }
    float* o = out + (size_t)f * 16 * 512 + 256 + c;
    for (int i = 0; i < 16; ++i) o[(size_t)i * 512] += (i == j1) ? m2 : m1;
}

// ---------------------------------------------------------------------------
extern "C" void kernel_launch(void* const* d_in, const int* in_sizes, int n_in,
                              void* d_out, int out_size, void* d_ws, size_t ws_size,
                              hipStream_t stream)
{
    const float* pf     = (const float*)d_in[0];   // 2048 x 2048
    const float* of     = (const float*)d_in[1];   // 6144 x 2048
    const float* Wp     = (const float*)d_in[2];
    const float* bp     = (const float*)d_in[3];
    const float* Wpr    = (const float*)d_in[4];
    const float* bpr    = (const float*)d_in[5];
    const float* Wo     = (const float*)d_in[6];
    const float* bo     = (const float*)d_in[7];
    const float* Wm_obj = (const float*)d_in[8];
    const float* bm_obj = (const float*)d_in[9];
    const float* Wm_per = (const float*)d_in[10];
    const float* bm_per = (const float*)d_in[11];
    float* out = (float*)d_out;                    // 2048 x 512

    char* ws = (char*)d_ws;
    size_t off = 0;
    auto alloc = [&](size_t bytes) {
        void* p = ws + off;
        off = (off + bytes + 255) & ~(size_t)255;
        return p;
    };
    unsigned short* W1t    = (unsigned short*)alloc((size_t)512 * 2048 * 2);
    unsigned short* Wot    = (unsigned short*)alloc((size_t)256 * 2048 * 2);
    unsigned short* WmTopT = (unsigned short*)alloc((size_t)512 * 256 * 2);
    unsigned short* WmOBot = (unsigned short*)alloc((size_t)256 * 256 * 2);
    unsigned short* WmPBot = (unsigned short*)alloc((size_t)256 * 256 * 2);
    float* bias1           = (float*)alloc(512 * 4);
    float* biasTop         = (float*)alloc(512 * 4);
    unsigned short* PAbf   = (unsigned short*)alloc((size_t)2048 * 512 * 2); // [P|A] bf16
    unsigned short* Obf    = (unsigned short*)alloc((size_t)6144 * 256 * 2);
    float* b_p             = (float*)alloc((size_t)2048 * 256 * 4);
    float* b_o             = (float*)alloc((size_t)6144 * 256 * 4);
    // Optional pre-converted bf16 activations (preferred path).
    unsigned short* pfbf   = (unsigned short*)alloc((size_t)2048 * 2048 * 2);
    unsigned short* ofbf   = (unsigned short*)alloc((size_t)6144 * 2048 * 2);
    const bool preconv = (ws_size >= off);     // deterministic host-side choice

    // 0) weight prep (transpose + bf16)
    prep_kernel<<<2048, 256, 0, stream>>>(Wp, bp, Wpr, bpr, Wo, Wm_obj, bm_obj,
                                          Wm_per, bm_per, W1t, Wot, WmTopT,
                                          WmOBot, WmPBot, bias1, biasTop);

    if (preconv) {
        // 0b) one streaming bf16 conversion pass; K-loops then run pure bf16.
        convert_bf16<<<2048, 256, 0, stream>>>(pf, pfbf, 2048 * 2048 / 8);
        convert_bf16<<<2048, 256, 0, stream>>>(of, ofbf, 6144 * 2048 / 8);
        // 1) PA = pf @ [Wp|Wpr] + [bp|bpr] -> bf16  (M=2048,N=512,K=2048)
        gemm_bf16_wmma<false><<<128, 256, 0, stream>>>(pfbf, 2048, W1t, bias1,
                                                       nullptr, PAbf, 512, 2048, 512, 2048);
        // 2) O = of @ Wo + bo -> bf16               (M=6144,N=256,K=2048)
        gemm_bf16_wmma<false><<<192, 256, 0, stream>>>(ofbf, 2048, Wot, bo,
                                                       nullptr, Obf, 256, 6144, 256, 2048);
    } else {
        // Fallback: convert A inline inside the K-loop.
        gemm_bf16_wmma<true><<<128, 256, 0, stream>>>(pf, 2048, W1t, bias1,
                                                      nullptr, PAbf, 512, 2048, 512, 2048);
        gemm_bf16_wmma<true><<<192, 256, 0, stream>>>(of, 2048, Wot, bo,
                                                      nullptr, Obf, 256, 6144, 256, 2048);
    }

    // 3) a_op = P @ [WmObjTop|WmPerTop] + [bm_obj|bm_per] -> f32 directly into d_out
    gemm_bf16_wmma<false><<<128, 256, 0, stream>>>(PAbf, 512, WmTopT, biasTop,
                                                   out, nullptr, 512, 2048, 512, 256);
    // 4) b_p = A @ WmPerBot -> f32                 (M=2048,N=256,K=256)
    gemm_bf16_wmma<false><<<64, 256, 0, stream>>>(PAbf + 256, 512, WmPBot, nullptr,
                                                  b_p, nullptr, 256, 2048, 256, 256);
    // 5) b_o = O @ WmObjBot -> f32                 (M=6144,N=256,K=256)
    gemm_bf16_wmma<false><<<192, 256, 0, stream>>>(Obf, 256, WmOBot, nullptr,
                                                   b_o, nullptr, 256, 6144, 256, 256);

    // 6) frame-wise max reductions folded into the output
    reduce_obj<<<128, 256, 0, stream>>>(b_o, out);
    reduce_per<<<128, 256, 0, stream>>>(b_p, out);
}
